// StructuralGNN_31576599560257
// MI455X (gfx1250) — compile-verified, matrix operand order checked by reading, not run on hardware
//
#include <hip/hip_runtime.h>
#include <hip/hip_bf16.h>
#include <math.h>

// ---------------------------------------------------------------------------
// CDNA5 (gfx1250) wave32 WMMA bf16 GNN forward pass.
// All dense math goes through v_wmma_f32_16x16x32_bf16.
// Edge aggregation (segment mean / edge softmax) uses L2-resident atomics.
// ---------------------------------------------------------------------------

typedef __attribute__((ext_vector_type(16))) __bf16 v16bf;
typedef __attribute__((ext_vector_type(8)))  float  v8f;

__device__ __forceinline__ v8f wmma_bf16(v16bf a, v16bf b, v8f c) {
  // 8 args: (neg_a, A, neg_b, B, c_mod, C, reuse_a, reuse_b)
  return __builtin_amdgcn_wmma_f32_16x16x32_bf16(false, a, false, b, (short)0, c,
                                                 false, false);
}

// A fragment (16x32, bf16) from row-major f32. Per lane: row = lane&15,
// koff = (lane>>4)*8; halves j<8 -> K=koff+j, j>=8 -> K=16+koff+(j-8).
__device__ __forceinline__ v16bf load_a_f32(const float* __restrict__ p, int koff) {
  v16bf a;
#pragma unroll
  for (int j = 0; j < 8; ++j) a[j]     = (__bf16)p[koff + j];
#pragma unroll
  for (int j = 0; j < 8; ++j) a[j + 8] = (__bf16)p[koff + 16 + j];
  return a;
}

__device__ __forceinline__ v16bf load_a_scaled(const float* __restrict__ p, int koff,
                                               float s) {
  v16bf a;
#pragma unroll
  for (int j = 0; j < 8; ++j) a[j]     = (__bf16)(p[koff + j] * s);
#pragma unroll
  for (int j = 0; j < 8; ++j) a[j + 8] = (__bf16)(p[koff + 16 + j] * s);
  return a;
}

// B fragment from pre-packed bf16 weights: one contiguous 32B load per lane.
__device__ __forceinline__ v16bf load_b_packed(const __bf16* __restrict__ packed,
                                               int t, int lane) {
  return *(const v16bf*)(packed + ((size_t)(t * 32 + lane) << 4));
}

__device__ __forceinline__ float lrelu(float x) { return x > 0.0f ? x : 0.2f * x; }

// Order-preserving float<->uint encode for atomicMax on floats.
__device__ __forceinline__ unsigned fenc(float f) {
  unsigned u = __float_as_uint(f);
  return (u & 0x80000000u) ? ~u : (u | 0x80000000u);
}
__device__ __forceinline__ float fdec(unsigned u) {
  unsigned v = (u & 0x80000000u) ? (u & 0x7fffffffu) : ~u;
  return __uint_as_float(v);
}

// ---------------------------------------------------------------------------
// Pack a row-major f32 weight [K, C] into bf16 B-fragment order:
// out[(((kt*ntiles)+nt)*32 + lane)*16 + j] = W[k(kt,lane,j), nt*16 + (lane&15)]
// ---------------------------------------------------------------------------
__global__ void pack_weight_kernel(const float* __restrict__ W, __bf16* __restrict__ out,
                                   int K, int C) {
  int idx = blockIdx.x * blockDim.x + threadIdx.x;
  int total = K * C;
  if (idx >= total) return;
  int j    = idx & 15;
  int lane = (idx >> 4) & 31;
  int t    = idx >> 9;
  int ntiles = C >> 4;
  int nt = t % ntiles;
  int kt = t / ntiles;
  int col  = nt * 16 + (lane & 15);
  int koff = (lane >> 4) << 3;
  int k = kt * 32 + (j < 8 ? (koff + j) : (16 + koff + (j - 8)));
  out[idx] = (__bf16)W[(size_t)k * C + col];
}

// ---------------------------------------------------------------------------
// Fused input transform: n2v projection + gated input projection.
// 1 block = 16 nodes, 4 waves = 4 column tiles of H=64.
// ---------------------------------------------------------------------------
__global__ __launch_bounds__(128)
void fused_input_kernel(const float* __restrict__ x,      // [N,256]
                        const float* __restrict__ n2v,    // [N,128]
                        const __bf16* __restrict__ p_n2v, // packed [128x64]
                        const __bf16* __restrict__ p_inp, // packed [320x64]
                        const __bf16* __restrict__ p_gate,// packed [320x64]
                        const float* __restrict__ n2v_b,
                        const float* __restrict__ inp_b,
                        const float* __restrict__ gate_b,
                        float* __restrict__ h0, int n) {
  __shared__ float lds_np[16 * 64];          // n2v_p tile [16,64]
  const int lane = threadIdx.x & 31;
  const int nt   = threadIdx.x >> 5;
  const int mrow = lane & 15;
  const int koff = (lane >> 4) << 3;
  const int tile = blockIdx.x;
  int r0 = tile * 16 + mrow; if (r0 >= n) r0 = n - 1;
  const int ncol = nt * 16 + mrow;

  // n2v_p = n2v @ Wn2v + b  (K = 128)
  v8f acc = {};
#pragma unroll
  for (int kt = 0; kt < 4; ++kt) {
    v16bf a = load_a_f32(n2v + (size_t)r0 * 128 + kt * 32, koff);
    v16bf b = load_b_packed(p_n2v, kt * 4 + nt, lane);
    acc = wmma_bf16(a, b, acc);
  }
  {
    const float bias = n2v_b[ncol];
#pragma unroll
    for (int v = 0; v < 8; ++v) {
      int m = v + ((lane >> 4) << 3);
      lds_np[m * 64 + ncol] = acc[v] + bias;   // C/D layout: M=v+8*(lane>>4), N=lane&15
    }
  }
  __syncthreads();

  // P = combined @ W_inp, G = combined @ W_gate; raw = x-part of P (K<256).
  v8f P = {}, G = {};
#pragma unroll
  for (int kt = 0; kt < 8; ++kt) {           // K = 0..255 from x
    v16bf a  = load_a_f32(x + (size_t)r0 * 256 + kt * 32, koff);
    v16bf bp = load_b_packed(p_inp,  kt * 4 + nt, lane);
    v16bf bgf = load_b_packed(p_gate, kt * 4 + nt, lane);
    P = wmma_bf16(a, bp, P);
    G = wmma_bf16(a, bgf, G);
  }
  v8f raw = P;                               // raw_proj = x @ W_inp[:256]
#pragma unroll
  for (int kt = 8; kt < 10; ++kt) {          // K = 256..319 from n2v_p (LDS)
    v16bf a  = load_a_f32(lds_np + mrow * 64 + (kt - 8) * 32, koff);
    v16bf bp = load_b_packed(p_inp,  kt * 4 + nt, lane);
    v16bf bgf = load_b_packed(p_gate, kt * 4 + nt, lane);
    P = wmma_bf16(a, bp, P);
    G = wmma_bf16(a, bgf, G);
  }

  const float bi  = inp_b[ncol];
  const float bgb = gate_b[ncol];
#pragma unroll
  for (int v = 0; v < 8; ++v) {
    int m = v + ((lane >> 4) << 3);
    int r = tile * 16 + m;
    if (r < n) {
      float xp = P[v] + bi;
      float rw = raw[v] + bi;
      float g  = 1.0f / (1.0f + expf(-(G[v] + bgb)));
      h0[(size_t)r * 64 + ncol] = g * xp + (1.0f - g) * rw;
    }
  }
}

// ---------------------------------------------------------------------------
// SAGE edge scatter: agg[dst] += h[src]; deg[dst] += 1.  L2-resident atomics.
// 1 thread = (edge, 16-feature chunk).
// ---------------------------------------------------------------------------
__global__ void sage_scatter_kernel(const int* __restrict__ ei, const float* __restrict__ h,
                                    float* __restrict__ agg, float* __restrict__ deg, int E_) {
  int idx = blockIdx.x * blockDim.x + threadIdx.x;
  int e = idx >> 2, part = idx & 3;
  if (e >= E_) return;
  int s = ei[e], d = ei[E_ + e];
  if (part == 0) atomicAdd(&deg[d], 1.0f);
  const float* hs = h + (size_t)s * 64;
  float* o = agg + (size_t)d * 64;
#pragma unroll
  for (int f = part * 16; f < part * 16 + 16; ++f) atomicAdd(&o[f], hs[f]);
}

// ---------------------------------------------------------------------------
// SAGE combine: out = (agg/deg) @ wl + bl + h @ wr, optional ReLU.
// ntiles waves; K=64 (2 WMMA k-steps per GEMM).
// ---------------------------------------------------------------------------
__global__ __launch_bounds__(128)
void sage_gemm_kernel(const float* __restrict__ agg, const float* __restrict__ deg,
                      const float* __restrict__ hin,
                      const __bf16* __restrict__ p_wl, const float* __restrict__ bl,
                      const __bf16* __restrict__ p_wr,
                      float* __restrict__ out, int ncols, int do_relu, int n) {
  const int lane = threadIdx.x & 31;
  const int nt   = threadIdx.x >> 5;
  const int ntiles = blockDim.x >> 5;
  const int mrow = lane & 15;
  const int koff = (lane >> 4) << 3;
  const int tile = blockIdx.x;
  int r0 = tile * 16 + mrow; if (r0 >= n) r0 = n - 1;
  const int ncol = nt * 16 + mrow;

  const float inv = 1.0f / fmaxf(deg[r0], 1.0f);
  v8f acc = {};
#pragma unroll
  for (int kt = 0; kt < 2; ++kt) {
    v16bf a = load_a_scaled(agg + (size_t)r0 * 64 + kt * 32, koff, inv);
    v16bf b = load_b_packed(p_wl, kt * ntiles + nt, lane);
    acc = wmma_bf16(a, b, acc);
  }
#pragma unroll
  for (int kt = 0; kt < 2; ++kt) {
    v16bf a = load_a_f32(hin + (size_t)r0 * 64 + kt * 32, koff);
    v16bf b = load_b_packed(p_wr, kt * ntiles + nt, lane);
    acc = wmma_bf16(a, b, acc);
  }
  const float bias = bl[ncol];
#pragma unroll
  for (int v = 0; v < 8; ++v) {
    int m = v + ((lane >> 4) << 3);
    int r = tile * 16 + m;
    if (r < n) {
      float val = acc[v] + bias;
      if (do_relu) val = fmaxf(val, 0.0f);
      out[(size_t)r * ncols + ncol] = val;
    }
  }
}

// ---------------------------------------------------------------------------
// Plain GEMM: out[N,ncols] = hin[N,64] @ W (for GAT xh). ncols/16 waves.
// ---------------------------------------------------------------------------
__global__ __launch_bounds__(256)
void gemm_h64_kernel(const float* __restrict__ hin, const __bf16* __restrict__ pw,
                     float* __restrict__ out, int ncols, int n) {
  const int lane = threadIdx.x & 31;
  const int nt   = threadIdx.x >> 5;
  const int ntiles = blockDim.x >> 5;
  const int mrow = lane & 15;
  const int koff = (lane >> 4) << 3;
  const int tile = blockIdx.x;
  int r0 = tile * 16 + mrow; if (r0 >= n) r0 = n - 1;
  const int ncol = nt * 16 + mrow;

  v8f acc = {};
#pragma unroll
  for (int kt = 0; kt < 2; ++kt) {
    v16bf a = load_a_f32(hin + (size_t)r0 * 64 + kt * 32, koff);
    v16bf b = load_b_packed(pw, kt * ntiles + nt, lane);
    acc = wmma_bf16(a, b, acc);
  }
#pragma unroll
  for (int v = 0; v < 8; ++v) {
    int m = v + ((lane >> 4) << 3);
    int r = tile * 16 + m;
    if (r < n) out[(size_t)r * ncols + ncol] = acc[v];
  }
}

// ---------------------------------------------------------------------------
// GAT attention scores + self-loop-seeded running max.
// ---------------------------------------------------------------------------
__global__ void gat_att_kernel(const float* __restrict__ xh, const float* __restrict__ att_src,
                               const float* __restrict__ att_dst, float* __restrict__ a_src,
                               float* __restrict__ a_dst, unsigned* __restrict__ emax_u,
                               float* __restrict__ esum, int n) {
  int id = blockIdx.x * blockDim.x + threadIdx.x;
  if (id >= 2 * n) return;
  int node = id >> 1, h = id & 1;
  const float* row = xh + (size_t)node * 128 + h * 64;
  const float* as = att_src + h * 64;
  const float* ad = att_dst + h * 64;
  float s = 0.f, d = 0.f;
#pragma unroll 8
  for (int k = 0; k < 64; ++k) { float v = row[k]; s += v * as[k]; d += v * ad[k]; }
  a_src[id] = s;
  a_dst[id] = d;
  emax_u[id] = fenc(lrelu(s + d));   // self-loop always exists -> seeds segment max
  esum[id] = 0.f;
}

__global__ void gat_edge_max_kernel(const int* __restrict__ ei, const float* __restrict__ a_src,
                                    const float* __restrict__ a_dst,
                                    unsigned* __restrict__ emax_u, int E_) {
  int e = blockIdx.x * blockDim.x + threadIdx.x;
  if (e >= E_) return;
  int s = ei[e], d = ei[E_ + e];
#pragma unroll
  for (int h = 0; h < 2; ++h) {
    float v = lrelu(a_src[s * 2 + h] + a_dst[d * 2 + h]);
    atomicMax(&emax_u[d * 2 + h], fenc(v));
  }
}

__global__ void gat_expsum_kernel(const int* __restrict__ ei, const float* __restrict__ a_src,
                                  const float* __restrict__ a_dst,
                                  const unsigned* __restrict__ emax_u,
                                  float* __restrict__ esum, int E_, int n) {
  int idx = blockIdx.x * blockDim.x + threadIdx.x;
  if (idx >= E_ + n) return;
  int s, d;
  if (idx < E_) { s = ei[idx]; d = ei[E_ + idx]; } else { s = d = idx - E_; }
#pragma unroll
  for (int h = 0; h < 2; ++h) {
    float v = lrelu(a_src[s * 2 + h] + a_dst[d * 2 + h]);
    atomicAdd(&esum[d * 2 + h], expf(v - fdec(emax_u[d * 2 + h])));
  }
}

__global__ void gat_scatter_kernel(const int* __restrict__ ei, const float* __restrict__ a_src,
                                   const float* __restrict__ a_dst,
                                   const unsigned* __restrict__ emax_u,
                                   const float* __restrict__ esum,
                                   const float* __restrict__ xh, float* __restrict__ gacc,
                                   int E_, int n) {
  int idx = blockIdx.x * blockDim.x + threadIdx.x;
  int e = idx >> 2, part = idx & 3;
  if (e >= E_ + n) return;
  int s, d;
  if (e < E_) { s = ei[e]; d = ei[E_ + e]; } else { s = d = e - E_; }
  float alpha[2];
#pragma unroll
  for (int h = 0; h < 2; ++h) {
    float v  = lrelu(a_src[s * 2 + h] + a_dst[d * 2 + h]);
    float ex = expf(v - fdec(emax_u[d * 2 + h]));
    alpha[h] = ex / (esum[d * 2 + h] + 1e-16f);
  }
  const float* x0 = xh + (size_t)s * 128;
  float* o = gacc + (size_t)d * 64;
#pragma unroll
  for (int f = part * 16; f < part * 16 + 16; ++f)
    atomicAdd(&o[f], alpha[0] * x0[f] + alpha[1] * x0[64 + f]);
}

__global__ void gat_finalize_kernel(const float* __restrict__ gacc,
                                    const float* __restrict__ gat_b,
                                    float* __restrict__ hout, int n) {
  int idx = blockIdx.x * blockDim.x + threadIdx.x;
  if (idx >= n * 64) return;
  int f = idx & 63;
  hout[idx] = fmaxf(0.5f * gacc[idx] + gat_b[f], 0.0f);   // mean over 2 heads + bias, ReLU
}

// ---------------------------------------------------------------------------
extern "C" void kernel_launch(void* const* d_in, const int* in_sizes, int n_in,
                              void* d_out, int out_size, void* d_ws, size_t ws_size,
                              hipStream_t stream) {
  (void)n_in; (void)out_size; (void)ws_size;
  const float* x       = (const float*)d_in[0];
  const int*   ei      = (const int*)d_in[1];
  const float* n2v     = (const float*)d_in[2];
  const float* n2v_w   = (const float*)d_in[3];
  const float* n2v_b   = (const float*)d_in[4];
  const float* inp_w   = (const float*)d_in[5];
  const float* inp_b   = (const float*)d_in[6];
  const float* gate_w  = (const float*)d_in[7];
  const float* gate_b  = (const float*)d_in[8];
  const float* s0_wl   = (const float*)d_in[9];
  const float* s0_bl   = (const float*)d_in[10];
  const float* s0_wr   = (const float*)d_in[11];
  const float* s1_wl   = (const float*)d_in[12];
  const float* s1_bl   = (const float*)d_in[13];
  const float* s1_wr   = (const float*)d_in[14];
  const float* gat_w   = (const float*)d_in[15];
  const float* att_src = (const float*)d_in[16];
  const float* att_dst = (const float*)d_in[17];
  const float* gat_b   = (const float*)d_in[18];
  const float* fin_wl  = (const float*)d_in[19];
  const float* fin_bl  = (const float*)d_in[20];
  const float* fin_wr  = (const float*)d_in[21];

  const int N = in_sizes[0] / 256;
  const int E = in_sizes[1] / 2;
  const int tiles = (N + 15) / 16;

  // ---- workspace carve (all L2-friendly f32 tables + packed bf16 weights) ----
  char* wsb = (char*)d_ws;
  size_t off = 0;
  auto carve = [&](size_t bytes) -> void* {
    void* p = wsb + off;
    off += (bytes + 255) & ~(size_t)255;
    return p;
  };
  float*    hA     = (float*)carve((size_t)N * 64 * 4);
  float*    hB     = (float*)carve((size_t)N * 64 * 4);
  float*    agg    = (float*)carve((size_t)N * 64 * 4);
  float*    gacc   = (float*)carve((size_t)N * 64 * 4);
  float*    xh     = (float*)carve((size_t)N * 128 * 4);
  float*    deg    = (float*)carve((size_t)N * 4);
  float*    a_src  = (float*)carve((size_t)N * 2 * 4);
  float*    a_dst  = (float*)carve((size_t)N * 2 * 4);
  unsigned* emax_u = (unsigned*)carve((size_t)N * 2 * 4);
  float*    esum   = (float*)carve((size_t)N * 2 * 4);
  __bf16*   p_n2v  = (__bf16*)carve(128 * 64 * 2);
  __bf16*   p_inp  = (__bf16*)carve(320 * 64 * 2);
  __bf16*   p_gate = (__bf16*)carve(320 * 64 * 2);
  __bf16*   p_s0wl = (__bf16*)carve(64 * 64 * 2);
  __bf16*   p_s0wr = (__bf16*)carve(64 * 64 * 2);
  __bf16*   p_s1wl = (__bf16*)carve(64 * 64 * 2);
  __bf16*   p_s1wr = (__bf16*)carve(64 * 64 * 2);
  __bf16*   p_gat  = (__bf16*)carve(64 * 128 * 2);
  __bf16*   p_fwl  = (__bf16*)carve(64 * 32 * 2);
  __bf16*   p_fwr  = (__bf16*)carve(64 * 32 * 2);

  auto pack = [&](const float* W, __bf16* dst, int K, int C) {
    int total = K * C;
    pack_weight_kernel<<<(total + 255) / 256, 256, 0, stream>>>(W, dst, K, C);
  };
  pack(n2v_w,  p_n2v,  128, 64);
  pack(inp_w,  p_inp,  320, 64);
  pack(gate_w, p_gate, 320, 64);
  pack(s0_wl,  p_s0wl, 64, 64);
  pack(s0_wr,  p_s0wr, 64, 64);
  pack(s1_wl,  p_s1wl, 64, 64);
  pack(s1_wr,  p_s1wr, 64, 64);
  pack(gat_w,  p_gat,  64, 128);
  pack(fin_wl, p_fwl,  64, 32);
  pack(fin_wr, p_fwr,  64, 32);

  // ---- fused input transform -> hA ----
  fused_input_kernel<<<tiles, 128, 0, stream>>>(x, n2v, p_n2v, p_inp, p_gate,
                                                n2v_b, inp_b, gate_b, hA, N);

  unsigned eg4 = (unsigned)(((size_t)E * 4 + 255) / 256);

  // ---- SAGE 0: hA -> hB (relu) ----
  hipMemsetAsync(agg, 0, (size_t)N * 64 * 4, stream);
  hipMemsetAsync(deg, 0, (size_t)N * 4, stream);
  sage_scatter_kernel<<<eg4, 256, 0, stream>>>(ei, hA, agg, deg, E);
  sage_gemm_kernel<<<tiles, 128, 0, stream>>>(agg, deg, hA, p_s0wl, s0_bl, p_s0wr,
                                              hB, 64, 1, N);

  // ---- SAGE 1: hB -> hA (relu) ----
  hipMemsetAsync(agg, 0, (size_t)N * 64 * 4, stream);
  hipMemsetAsync(deg, 0, (size_t)N * 4, stream);
  sage_scatter_kernel<<<eg4, 256, 0, stream>>>(ei, hB, agg, deg, E);
  sage_gemm_kernel<<<tiles, 128, 0, stream>>>(agg, deg, hB, p_s1wl, s1_bl, p_s1wr,
                                              hA, 64, 1, N);

  // ---- GAT: hA -> hB (relu) ----
  gemm_h64_kernel<<<tiles, 256, 0, stream>>>(hA, p_gat, xh, 128, N);
  gat_att_kernel<<<(2 * N + 255) / 256, 256, 0, stream>>>(xh, att_src, att_dst,
                                                          a_src, a_dst, emax_u, esum, N);
  gat_edge_max_kernel<<<(E + 255) / 256, 256, 0, stream>>>(ei, a_src, a_dst, emax_u, E);
  gat_expsum_kernel<<<(E + N + 255) / 256, 256, 0, stream>>>(ei, a_src, a_dst, emax_u,
                                                             esum, E, N);
  hipMemsetAsync(gacc, 0, (size_t)N * 64 * 4, stream);
  unsigned esg4 = (unsigned)(((size_t)(E + N) * 4 + 255) / 256);
  gat_scatter_kernel<<<esg4, 256, 0, stream>>>(ei, a_src, a_dst, emax_u, esum,
                                               xh, gacc, E, N);
  gat_finalize_kernel<<<((size_t)N * 64 + 255) / 256, 256, 0, stream>>>(gacc, gat_b, hB, N);

  // ---- final SAGE: hB -> d_out [N,32] ----
  hipMemsetAsync(agg, 0, (size_t)N * 64 * 4, stream);
  hipMemsetAsync(deg, 0, (size_t)N * 4, stream);
  sage_scatter_kernel<<<eg4, 256, 0, stream>>>(ei, hB, agg, deg, E);
  sage_gemm_kernel<<<tiles, 64, 0, stream>>>(agg, deg, hB, p_fwl, fin_bl, p_fwr,
                                             (float*)d_out, 32, 0, N);
}
